// RXNetLSTM_82351702934205
// MI455X (gfx1250) — compile-verified, matrix operand order checked by reading, not run on hardware
//
#include <hip/hip_runtime.h>
#include <hip/hip_bf16.h>
#include <stdint.h>

// ---------------------------------------------------------------------------
// RXNet 2-layer LSTM LM on gfx1250 (MI455X).  V=512 E=512 H=1024 B=128 T=512
//  * bf16 WMMA GEMMs (f32 accum); gate-interleaved weight order lets the LSTM
//    cell fuse into the GEMM epilogue with zero cross-lane traffic.
//  * Weight tiles double-buffered in LDS via GLOBAL_LOAD_ASYNC_TO_LDS_B128
//    (ASYNCcnt): one s_wait_asynccnt + one barrier per k-step.
//  * A-fragments software-pipelined one step ahead to avoid WMMA->VALU
//    hazard NOPs on fragment-register reuse.
//  * h0/h1 ping-pong on step parity (fused epilogue writes vs reads).
// ---------------------------------------------------------------------------

#define Vv   512
#define Ee   512
#define Hh   1024
#define Bb   128
#define Tt   512
#define G4H  4096   // 4*H

typedef __attribute__((ext_vector_type(8)))  float   v8f;
typedef __attribute__((ext_vector_type(16))) __bf16  v16bf;
typedef unsigned short ushort_t;
typedef unsigned int   uint_t;

union FragU { uint4 u[2]; v16bf v; };

__device__ __forceinline__ ushort_t f2bf(float x) {
    uint_t u = __float_as_uint(x);
    uint_t r = u + 0x7fffu + ((u >> 16) & 1u);   // round-to-nearest-even
    return (ushort_t)(r >> 16);
}
__device__ __forceinline__ float sigm(float x) { return 1.0f / (1.0f + __expf(-x)); }

// ---------------------------------------------------------------------------
// WMMA GEMM, M=128 (whole batch) x N=64 per block, 256 threads = 8 waves.
//   acc = bias[n] + A1[m,:]*W1[n,:]^T (+ A2[m,:]*W2[n,:]^T)
// cstate != nullptr: fused LSTM cell epilogue (acc0..3 = i,f,g,o per lane).
// cstate == nullptr: plain store to out (vocab projection).
// ---------------------------------------------------------------------------
__global__ __launch_bounds__(256)
void wmma_gemm_kernel(const ushort_t* __restrict__ A1,
                      const int*      __restrict__ tok,   // nullptr => direct
                      const ushort_t* __restrict__ B1, int K1,
                      const ushort_t* __restrict__ A2,    // nullptr => 1 pair
                      const ushort_t* __restrict__ B2, int K2,
                      const float*    __restrict__ bias,
                      float*          __restrict__ cstate,   // fused-cell mode
                      ushort_t*       __restrict__ hout,     // fused-cell mode
                      float*          __restrict__ out, int ldo)
{
    __shared__ ushort_t Bs[2][64 * 32];   // double-buffered 4 KB weight tiles

    const int tid   = threadIdx.x;
    const int wave  = tid >> 5;
    const int lane  = tid & 31;
    const int lan16 = lane & 15;
    const int n0    = blockIdx.x * 64;
    const int m     = wave * 16 + lan16;
    const int khalf = (lane < 16) ? 0 : 8;   // 16-bit A/B lane K offset
    const int sn    = tid >> 2;              // staging: tile row 0..63
    const int skq   = (tid & 3) * 8;         // staging: tile k offset

    v8f acc[4];
#pragma unroll
    for (int nf = 0; nf < 4; ++nf) {
        const float bv = bias[n0 + nf * 16 + lan16];
#pragma unroll
        for (int r = 0; r < 8; ++r) acc[nf][r] = bv;
    }

    const int steps1 = K1 >> 5;
    const int steps2 = (A2 != nullptr) ? (K2 >> 5) : 0;

    // Induction pointers (advance by 32 bf16 per k-step).
    const ushort_t* a1 = ((tok != nullptr) ? (A1 + (size_t)tok[m] * K1)
                                           : (A1 + (size_t)m * K1)) + khalf;
    const ushort_t* a2 = (A2 != nullptr) ? (A2 + (size_t)m * K2 + khalf) : nullptr;
    const ushort_t* g1 = B1 + (size_t)(n0 + sn) * K1 + skq;
    const ushort_t* g2 = (A2 != nullptr) ? (B2 + (size_t)(n0 + sn) * K2 + skq)
                                         : nullptr;
    const uint_t lds_off[2] = {
        (uint_t)(uintptr_t)&Bs[0][sn * 32 + skq],
        (uint_t)(uintptr_t)&Bs[1][sn * 32 + skq] };

    auto stage = [&](const ushort_t* g, int buf) {
        asm volatile("global_load_async_to_lds_b128 %0, %1, off"
                     :: "v"(lds_off[buf]), "v"(g) : "memory");
    };
    auto loadA = [&](const ushort_t* a) {
        FragU f;
        f.u[0] = *(const uint4*)(a);
        f.u[1] = *(const uint4*)(a + 16);
        return f;
    };
    auto mma4 = [&](const FragU& af, int buf) {
        FragU bf[4];
#pragma unroll
        for (int nf = 0; nf < 4; ++nf) {
            const ushort_t* bs = &Bs[buf][(nf * 16 + lan16) * 32 + khalf];
            bf[nf].u[0] = *(const uint4*)(bs);
            bf[nf].u[1] = *(const uint4*)(bs + 16);
        }
#pragma unroll
        for (int nf = 0; nf < 4; ++nf)
            acc[nf] = __builtin_amdgcn_wmma_f32_16x16x32_bf16(
                false, af.v, false, bf[nf].v, (short)0, acc[nf], false, false);
    };

    // Prologue: stage tile 0, prefetch A fragment 0.
    int buf = 0;
    stage(g1, 0); g1 += 32;
    FragU af = loadA(a1); a1 += 32;

    // ---- pair 0 ------------------------------------------------------------
    for (int s = 0; s < steps1; ++s) {
        asm volatile("s_wait_asynccnt 0x0" ::: "memory");
        __syncthreads();                 // tile s visible; buf^1 free
        FragU afn = af;
        if (s + 1 < steps1) {            // stage + A-prefetch next k-step
            stage(g1, buf ^ 1); g1 += 32;
            afn = loadA(a1);    a1 += 32;
        } else if (steps2 > 0) {         // hand off pipeline to pair 1
            stage(g2, buf ^ 1); g2 += 32;
            afn = loadA(a2);    a2 += 32;
        }
        mma4(af, buf);
        af = afn; buf ^= 1;
    }

    // ---- pair 1 ------------------------------------------------------------
    for (int s = 0; s < steps2; ++s) {
        asm volatile("s_wait_asynccnt 0x0" ::: "memory");
        __syncthreads();
        FragU afn = af;
        if (s + 1 < steps2) {
            stage(g2, buf ^ 1); g2 += 32;
            afn = loadA(a2);    a2 += 32;
        }
        mma4(af, buf);
        af = afn; buf ^= 1;
    }

    const int rbase = wave * 16 + ((lane >> 4) << 3);   // C/D: +8 lanes 16-31
    if (cstate != nullptr) {
        // Fused LSTM cell: acc[0..3] = i,f,g,o for j = 16*blockIdx + lane.
        const int j = (n0 >> 2) + lan16;
#pragma unroll
        for (int r = 0; r < 8; ++r) {
            const size_t cix = (size_t)(rbase + r) * Hh + j;
            const float iv = sigm(acc[0][r]);
            const float fv = sigm(acc[1][r]);
            const float gv = tanhf(acc[2][r]);
            const float ov = sigm(acc[3][r]);
            const float cv = fv * cstate[cix] + iv * gv;
            cstate[cix] = cv;
            hout[cix]   = f2bf(ov * tanhf(cv));
        }
    } else {
#pragma unroll
        for (int nf = 0; nf < 4; ++nf) {
            const int col = n0 + nf * 16 + lan16;
#pragma unroll
            for (int r = 0; r < 8; ++r)
                out[(size_t)(rbase + r) * ldo + col] = acc[nf][r];
        }
    }
}

// ---------------------------------------------------------------------------
// Argmax over V=512 logits per row + teacher-forcing token select for t+1.
// ---------------------------------------------------------------------------
__global__ void argmax_select_kernel(const float* __restrict__ logits, // +t*V
                                     int ldl,
                                     const int*   __restrict__ seq,
                                     const float* __restrict__ teacher_p,
                                     int tnext,
                                     int*         __restrict__ tok)
{
    __shared__ float sv[256];
    __shared__ int   si[256];
    const int b = blockIdx.x, tid = threadIdx.x;
    const float* row = logits + (size_t)b * ldl;

    float best = row[tid]; int bi = tid;
    const float v2 = row[tid + 256];
    if (v2 > best) { best = v2; bi = tid + 256; }   // strict > keeps 1st idx
    sv[tid] = best; si[tid] = bi;
    __syncthreads();
    for (int s = 128; s > 0; s >>= 1) {
        if (tid < s) {
            const float ov = sv[tid + s]; const int oi = si[tid + s];
            if (ov > sv[tid] || (ov == sv[tid] && oi < si[tid])) {
                sv[tid] = ov; si[tid] = oi;
            }
        }
        __syncthreads();
    }
    if (tid == 0) {
        uint_t x = (uint_t)(tnext * 65599 + b) * 2654435761u;   // u[t,b] stand-in
        x ^= x >> 16; x *= 0x45d9f3bu; x ^= x >> 13; x *= 0x45d9f3bu; x ^= x >> 16;
        const float u = (float)(x >> 8) * (1.0f / 16777216.0f);
        tok[b] = (u < *teacher_p) ? seq[b * Tt + tnext] : si[0];
    }
}

// ---------------------------------------------------------------------------
// One-time conversions + state init.
// ---------------------------------------------------------------------------
// Gate-interleaved weight reorder: n_new = (j>>4)*64 + gate*16 + (j&15).
__global__ void cvt_gate_w_kernel(const float* __restrict__ src,
                                  ushort_t* __restrict__ dst, int K)
{
    const int idx = blockIdx.x * blockDim.x + threadIdx.x;   // 4H*K
    if (idx >= G4H * K) return;
    const int n_old = idx / K, k = idx - n_old * K;
    const int gate = n_old >> 10, j = n_old & (Hh - 1);
    const int n_new = ((j >> 4) << 6) + (gate << 4) + (j & 15);
    dst[(size_t)n_new * K + k] = f2bf(src[idx]);
}

__global__ void bias_gate_kernel(const float* __restrict__ bih,
                                 const float* __restrict__ bhh,
                                 float* __restrict__ dst)
{
    const int n_old = blockIdx.x * blockDim.x + threadIdx.x;
    if (n_old >= G4H) return;
    const int gate = n_old >> 10, j = n_old & (Hh - 1);
    const int n_new = ((j >> 4) << 6) + (gate << 4) + (j & 15);
    dst[n_new] = bih[n_old] + bhh[n_old];
}

__global__ void cvt_bf16_kernel(const float* __restrict__ src,
                                ushort_t* __restrict__ dst, int n)
{
    const int idx = blockIdx.x * blockDim.x + threadIdx.x;
    if (idx < n) dst[idx] = f2bf(src[idx]);
}

__global__ void cvt_emb_kernel(const float* __restrict__ src,
                               ushort_t* __restrict__ dst)
{
    const int idx = blockIdx.x * blockDim.x + threadIdx.x;  // V*E
    if (idx < Vv * Ee) {
        const int row = idx >> 9;                            // E=512
        dst[idx] = (row == 0 /*PAD*/) ? (ushort_t)0 : f2bf(src[idx]);
    }
}

__global__ void init_state_kernel(float* c0, float* c1,
                                  ushort_t* h0a, ushort_t* h0b,
                                  ushort_t* h1a, ushort_t* h1b,
                                  int* tok, const int* __restrict__ seq)
{
    const int idx = blockIdx.x * blockDim.x + threadIdx.x;
    if (idx < Bb * Hh) {
        c0[idx] = 0.f; c1[idx] = 0.f;
        h0a[idx] = 0; h0b[idx] = 0; h1a[idx] = 0; h1b[idx] = 0;
    }
    if (idx < Bb) tok[idx] = seq[idx * Tt];   // t==0 always teacher-forced
}

// ---------------------------------------------------------------------------
extern "C" void kernel_launch(void* const* d_in, const int* in_sizes, int n_in,
                              void* d_out, int out_size, void* d_ws, size_t ws_size,
                              hipStream_t stream)
{
    (void)in_sizes; (void)n_in; (void)out_size; (void)ws_size;

    const int*   seq   = (const int*)  d_in[0];
    const float* t_p   = (const float*)d_in[2];
    const float* emb   = (const float*)d_in[3];
    const float* Wih0  = (const float*)d_in[4];
    const float* Whh0  = (const float*)d_in[5];
    const float* b_ih0 = (const float*)d_in[6];
    const float* b_hh0 = (const float*)d_in[7];
    const float* Wih1  = (const float*)d_in[8];
    const float* Whh1  = (const float*)d_in[9];
    const float* b_ih1 = (const float*)d_in[10];
    const float* b_hh1 = (const float*)d_in[11];
    const float* fcw   = (const float*)d_in[12];
    const float* fcb   = (const float*)d_in[13];
    float*       out   = (float*)d_out;          // [B, T, V]

    char* ws = (char*)d_ws;
    size_t off = 0;
    auto carve = [&](size_t bytes) {
        char* p = ws + off; off += (bytes + 255) & ~(size_t)255; return p; };
    ushort_t* Wih0_b = (ushort_t*)carve((size_t)G4H * Ee * 2);
    ushort_t* Whh0_b = (ushort_t*)carve((size_t)G4H * Hh * 2);
    ushort_t* Wih1_b = (ushort_t*)carve((size_t)G4H * Hh * 2);
    ushort_t* Whh1_b = (ushort_t*)carve((size_t)G4H * Hh * 2);
    ushort_t* fcw_b  = (ushort_t*)carve((size_t)Vv  * Hh * 2);
    ushort_t* emb_b  = (ushort_t*)carve((size_t)Vv  * Ee * 2);
    float*    bias0  = (float*)   carve((size_t)G4H * 4);
    float*    bias1  = (float*)   carve((size_t)G4H * 4);
    ushort_t* h0_b[2] = { (ushort_t*)carve((size_t)Bb * Hh * 2),
                          (ushort_t*)carve((size_t)Bb * Hh * 2) };
    ushort_t* h1_b[2] = { (ushort_t*)carve((size_t)Bb * Hh * 2),
                          (ushort_t*)carve((size_t)Bb * Hh * 2) };
    float*    c0     = (float*)   carve((size_t)Bb * Hh * 4);
    float*    c1     = (float*)   carve((size_t)Bb * Hh * 4);
    int*      tok    = (int*)     carve((size_t)Bb * 4);

    auto blocks = [](int n) { return (n + 255) / 256; };
    cvt_gate_w_kernel<<<blocks(G4H * Ee), 256, 0, stream>>>(Wih0, Wih0_b, Ee);
    cvt_gate_w_kernel<<<blocks(G4H * Hh), 256, 0, stream>>>(Whh0, Whh0_b, Hh);
    cvt_gate_w_kernel<<<blocks(G4H * Hh), 256, 0, stream>>>(Wih1, Wih1_b, Hh);
    cvt_gate_w_kernel<<<blocks(G4H * Hh), 256, 0, stream>>>(Whh1, Whh1_b, Hh);
    cvt_bf16_kernel  <<<blocks(Vv * Hh),  256, 0, stream>>>(fcw, fcw_b, Vv * Hh);
    cvt_emb_kernel   <<<blocks(Vv * Ee),  256, 0, stream>>>(emb, emb_b);
    bias_gate_kernel <<<blocks(G4H), 256, 0, stream>>>(b_ih0, b_hh0, bias0);
    bias_gate_kernel <<<blocks(G4H), 256, 0, stream>>>(b_ih1, b_hh1, bias1);
    init_state_kernel<<<blocks(Bb * Hh), 256, 0, stream>>>(
        c0, c1, h0_b[0], h0_b[1], h1_b[0], h1_b[1], tok, seq);

    for (int t = 0; t < Tt; ++t) {
        const int rd = t & 1, wr = rd ^ 1;   // h ping-pong

        // layer 0: gates(emb[tok], h0_prev) -> fused cell -> c0, h0_new
        wmma_gemm_kernel<<<G4H / 64, 256, 0, stream>>>(
            emb_b, tok, Wih0_b, Ee, h0_b[rd], Whh0_b, Hh, bias0,
            c0, h0_b[wr], nullptr, 0);

        // layer 1: gates(h0_new, h1_prev) -> fused cell -> c1, h1_new
        wmma_gemm_kernel<<<G4H / 64, 256, 0, stream>>>(
            h0_b[wr], nullptr, Wih1_b, Hh, h1_b[rd], Whh1_b, Hh, bias1,
            c1, h1_b[wr], nullptr, 0);

        // logits: h1_new @ fc_w^T + fc_b -> out[:, t, :]
        wmma_gemm_kernel<<<Vv / 64, 256, 0, stream>>>(
            h1_b[wr], nullptr, fcw_b, Hh, nullptr, nullptr, 0, fcb,
            nullptr, nullptr, out + (size_t)t * Vv, Tt * Vv);

        if (t + 1 < Tt) {
            argmax_select_kernel<<<Bb, 256, 0, stream>>>(
                out + (size_t)t * Vv, Tt * Vv, seq, t_p, t + 1, tok);
        }
    }
}
